// SelfAttention2D_33011118637366
// MI455X (gfx1250) — matrix operand order repaired and checked by reading, not compiled, not run on hardware
//
#include <hip/hip_runtime.h>

typedef __attribute__((ext_vector_type(16))) _Float16 v16h;
typedef __attribute__((ext_vector_type(8)))  float    v8f;

#define CIN   64
#define COUT  64
#define DH    24
#define DPAD  32
#define BATCH 4
#define NTOK  9216   /* 96*96 */
#define PSTR  72     /* padded row stride (halves) of P tile: 144B = 36 banks -> conflict-free */

// 1/sqrt(24) * log2(e): folded into Q so softmax uses exp2 (v_exp_f32).
#define QSCALE (0.2041241452319315f * 1.4426950408889634f)

// Row-max butterfly over each 16-lane half, one VALU op per step:
// DPP is an operand modifier, so max(shuffle(x), x) is a single
// v_max_num_f32_dpp.  xor 1,2,4 via DPP8; xor 8 via DPP16 row_ror:8
// (rotation by 8 within a 16-lane row == xor-8).
__device__ __forceinline__ float red_max16(float x) {
    float y;
    asm("v_max_num_f32_dpp %0, %1, %1 dpp8:[1,0,3,2,5,4,7,6]" : "=v"(y) : "v"(x));
    asm("v_max_num_f32_dpp %0, %1, %1 dpp8:[2,3,0,1,6,7,4,5]" : "=v"(x) : "v"(y));
    asm("v_max_num_f32_dpp %0, %1, %1 dpp8:[4,5,6,7,0,1,2,3]" : "=v"(y) : "v"(x));
    asm("v_max_num_f32_dpp %0, %1, %1 row_ror:8 row_mask:0xf bank_mask:0xf" : "=v"(x) : "v"(y));
    return x;
}

// max3 without LLVM's canonicalize ops.
__device__ __forceinline__ float max3f(float a, float b, float c) {
    float d;
    asm("v_max3_num_f32 %0, %1, %2, %3" : "=v"(d) : "v"(a), "v"(b), "v"(c));
    return d;
}

// Raw v_exp_f32: inputs are always <= 0 here; underflow-to-zero is desired.
__device__ __forceinline__ float fast_exp2(float x) {
    return __builtin_amdgcn_exp2f(x);
}

// Broadcast from lane (lane&16)|8 within each 16-lane half (epilogue only).
__device__ __forceinline__ float bcast_lane8(float x) {
    return __int_as_float(__builtin_amdgcn_ds_swizzle(__float_as_int(x), 0x0110));
}

// ---------------------------------------------------------------------------
// Phase 1: QKV projections (1x1 convs), convert to f16 in WMMA-friendly
// layouts.  Qh/Kh: [B][N][DPAD] row-major (zero-padded d>=24).
//           Vh:    [B][DPAD][N] d-major; pad row d==24 set to 1.0 so the
//                  P*V WMMA also produces the softmax row-sum (denominator)
//                  in output column d=24; rows 25..31 are zero.
// ---------------------------------------------------------------------------
__global__ __launch_bounds__(256) void qkv_proj_kernel(
    const float* __restrict__ x,
    const float* __restrict__ wq, const float* __restrict__ bq,
    const float* __restrict__ wk, const float* __restrict__ bk,
    const float* __restrict__ wv, const float* __restrict__ bv,
    _Float16* __restrict__ Qh, _Float16* __restrict__ Kh,
    _Float16* __restrict__ Vh)
{
    __shared__ float sW[3 * DH * CIN + 3 * DH];
    float* sWq = sW;
    float* sWk = sW + DH * CIN;
    float* sWv = sW + 2 * DH * CIN;
    float* sBq = sW + 3 * DH * CIN;
    float* sBk = sBq + DH;
    float* sBv = sBk + DH;
    for (int t = threadIdx.x; t < DH * CIN; t += 256) {
        sWq[t] = wq[t]; sWk[t] = wk[t]; sWv[t] = wv[t];
    }
    if (threadIdx.x < DH) {
        sBq[threadIdx.x] = bq[threadIdx.x];
        sBk[threadIdx.x] = bk[threadIdx.x];
        sBv[threadIdx.x] = bv[threadIdx.x];
    }
    __syncthreads();

    const int gid = blockIdx.x * 256 + threadIdx.x;   // over B*N
    const int b = gid / NTOK;
    const int i = gid - b * NTOK;

    float xc[CIN];
    const float* xp = x + (size_t)b * CIN * NTOK + i;
#pragma unroll
    for (int c = 0; c < CIN; ++c) xc[c] = xp[(size_t)c * NTOK];

    __attribute__((aligned(16))) _Float16 qrow[DPAD];
    __attribute__((aligned(16))) _Float16 krow[DPAD];
#pragma unroll
    for (int d = DH; d < DPAD; ++d) { qrow[d] = (_Float16)0.f; krow[d] = (_Float16)0.f; }

    _Float16* vcol = Vh + (size_t)b * DPAD * NTOK + i;
    for (int d = 0; d < DH; ++d) {
        float aq = sBq[d], ak = sBk[d], av = sBv[d];
        const float* wqd = sWq + d * CIN;
        const float* wkd = sWk + d * CIN;
        const float* wvd = sWv + d * CIN;
#pragma unroll
        for (int c = 0; c < CIN; ++c) {
            aq += wqd[c] * xc[c];
            ak += wkd[c] * xc[c];
            av += wvd[c] * xc[c];
        }
        qrow[d] = (_Float16)(aq * QSCALE);
        krow[d] = (_Float16)ak;
        vcol[(size_t)d * NTOK] = (_Float16)av;
    }
    vcol[(size_t)DH * NTOK] = (_Float16)1.f;   // ones row -> row-sum column
#pragma unroll
    for (int d = DH + 1; d < DPAD; ++d) vcol[(size_t)d * NTOK] = (_Float16)0.f;

    float4* qdst = (float4*)(Qh + ((size_t)b * NTOK + i) * DPAD);
    float4* kdst = (float4*)(Kh + ((size_t)b * NTOK + i) * DPAD);
    const float4* qs = (const float4*)qrow;
    const float4* ks = (const float4*)krow;
#pragma unroll
    for (int t = 0; t < 4; ++t) { qdst[t] = qs[t]; kdst[t] = ks[t]; }
}

// ---------------------------------------------------------------------------
// Phase 2: flash attention.  One wave per 16-query tile, streaming keys in
// chunks of 64 with online softmax.  8x v_wmma_f32_16x16x32_f16 per chunk.
// The softmax denominator is accumulated by the WMMA itself (V ones-row).
//   attn out: [B][DPAD][N] f32 (rows d>=24 unused)
// ---------------------------------------------------------------------------
__global__ __launch_bounds__(256) void flash_attn_kernel(
    const _Float16* __restrict__ Qh, const _Float16* __restrict__ Kh,
    const _Float16* __restrict__ Vh, float* __restrict__ attn)
{
    __shared__ __attribute__((aligned(16))) _Float16 Plds[8][16 * PSTR]; // 2304B/wave

    const int lane = threadIdx.x & 31;
    const int wid  = threadIdx.x >> 5;
    const int hi   = lane >> 4;     // lane half: 0 or 1
    const int ln   = lane & 15;

    const int qt = blockIdx.x * 8 + wid;       // 16-query tile id
    const int b  = qt / (NTOK / 16);
    const int i0 = (qt - b * (NTOK / 16)) * 16;

    // Q as A-fragment (16x32 f16): lane = row M=ln,
    // elems 0..15 = d hi*16+{0..15} (contiguous).
    v16h aq;
    {
        const _Float16* qp = Qh + ((size_t)b * NTOK + i0 + ln) * DPAD + hi * 16;
        ((float4*)&aq)[0] = ((const float4*)qp)[0];
        ((float4*)&aq)[1] = ((const float4*)(qp + 8))[0];
    }

    v8f O0 = {};  // out dims 0..15
    v8f O1 = {};  // out dims 16..31; col d=24 = softmax denominator
    float m[8];
#pragma unroll
    for (int r = 0; r < 8; ++r) m[r] = -1e30f;

    _Float16* Pw = &Plds[wid][0];
    const _Float16* kbase = Kh + (size_t)b * NTOK * DPAD;
    const _Float16* vbase = Vh + (size_t)b * DPAD * NTOK;

    for (int j0 = 0; j0 < NTOK; j0 += 64) {
        // K as B-fragments (32x16): lane = column j, contiguous d = hi*16+0..15
        v16h bk0, bk1, bk2, bk3;
        {
            const _Float16* kp = kbase + (size_t)(j0 + ln) * DPAD + hi * 16;
            ((float4*)&bk0)[0] = ((const float4*)kp)[0];
            ((float4*)&bk0)[1] = ((const float4*)(kp + 8))[0];
            ((float4*)&bk1)[0] = ((const float4*)(kp + 16 * DPAD))[0];
            ((float4*)&bk1)[1] = ((const float4*)(kp + 16 * DPAD + 8))[0];
            ((float4*)&bk2)[0] = ((const float4*)(kp + 32 * DPAD))[0];
            ((float4*)&bk2)[1] = ((const float4*)(kp + 32 * DPAD + 8))[0];
            ((float4*)&bk3)[0] = ((const float4*)(kp + 48 * DPAD))[0];
            ((float4*)&bk3)[1] = ((const float4*)(kp + 48 * DPAD + 8))[0];
        }
        // V as B-fragments: lane = column d, contiguous j = j0+js*32+hi*16+0..15
        v16h bv00, bv10, bv01, bv11;   // bv[js][dt]
        {
            const _Float16* vp = vbase + (size_t)ln * NTOK + j0 + hi * 16;
            ((float4*)&bv00)[0] = ((const float4*)vp)[0];
            ((float4*)&bv00)[1] = ((const float4*)(vp + 8))[0];
            ((float4*)&bv10)[0] = ((const float4*)(vp + 32))[0];
            ((float4*)&bv10)[1] = ((const float4*)(vp + 40))[0];
            ((float4*)&bv01)[0] = ((const float4*)(vp + (size_t)16 * NTOK))[0];
            ((float4*)&bv01)[1] = ((const float4*)(vp + (size_t)16 * NTOK + 8))[0];
            ((float4*)&bv11)[0] = ((const float4*)(vp + (size_t)16 * NTOK + 32))[0];
            ((float4*)&bv11)[1] = ((const float4*)(vp + (size_t)16 * NTOK + 40))[0];
        }

        // Scores (scale folded into Q, log2 domain).  4 independent WMMAs.
        v8f S0 = {}, S1 = {}, S2 = {}, S3 = {};
        S0 = __builtin_amdgcn_wmma_f32_16x16x32_f16(false, aq, false, bk0, (short)0, S0, false, false);
        S1 = __builtin_amdgcn_wmma_f32_16x16x32_f16(false, aq, false, bk1, (short)0, S1, false, false);
        S2 = __builtin_amdgcn_wmma_f32_16x16x32_f16(false, aq, false, bk2, (short)0, S2, false, false);
        S3 = __builtin_amdgcn_wmma_f32_16x16x32_f16(false, aq, false, bk3, (short)0, S3, false, false);

        // Online softmax: row r of this lane's half is query i0 + r + 8*hi.
        // m[r] is lane-replicated within the half, so it can be folded into
        // the pre-butterfly max: butterfly(max(S..., m)) == max(m, rowmax).
        // No explicit row-sum: the ones-row of V accumulates it in O1[d=24].
#pragma unroll
        for (int r = 0; r < 8; ++r) {
            float t = max3f(S0[r], S1[r], S2[r]);
            t = max3f(t, S3[r], m[r]);
            const float mn    = red_max16(t);
            const float alpha = fast_exp2(m[r] - mn);
            m[r] = mn;
            const float p0 = fast_exp2(S0[r] - mn);
            const float p1 = fast_exp2(S1[r] - mn);
            const float p2 = fast_exp2(S2[r] - mn);
            const float p3 = fast_exp2(S3[r] - mn);
            O0[r] *= alpha;
            O1[r] *= alpha;
            // Stage P (f16) row-major [q][j] (padded stride) for A-frag reload.
            _Float16* pr = Pw + (r + 8 * hi) * PSTR;
            pr[ln]      = (_Float16)p0;
            pr[16 + ln] = (_Float16)p1;
            pr[32 + ln] = (_Float16)p2;
            pr[48 + ln] = (_Float16)p3;
        }

        // Reload P as two A-fragments (same-wave DS ordering; no barrier).
        v16h ap0, ap1;
        const _Float16* pp = Pw + ln * PSTR + hi * 16;
        ((float4*)&ap0)[0] = ((const float4*)pp)[0];
        ((float4*)&ap0)[1] = ((const float4*)(pp + 8))[0];
        ((float4*)&ap1)[0] = ((const float4*)(pp + 32))[0];
        ((float4*)&ap1)[1] = ((const float4*)(pp + 40))[0];

        O0 = __builtin_amdgcn_wmma_f32_16x16x32_f16(false, ap0, false, bv00, (short)0, O0, false, false);
        O0 = __builtin_amdgcn_wmma_f32_16x16x32_f16(false, ap1, false, bv10, (short)0, O0, false, false);
        O1 = __builtin_amdgcn_wmma_f32_16x16x32_f16(false, ap0, false, bv01, (short)0, O1, false, false);
        O1 = __builtin_amdgcn_wmma_f32_16x16x32_f16(false, ap1, false, bv11, (short)0, O1, false, false);
    }

    // Normalize and write attn[b][d][i]  (C/D layout: row = r+8*hi, col = ln).
    // Row denominator sits in O1[r] at lane (hi*16+8) (V ones-row, d=24).
    float* abase = attn + (size_t)b * DPAD * NTOK + i0;
#pragma unroll
    for (int r = 0; r < 8; ++r) {
        const float lsum = bcast_lane8(O1[r]);
        const float inv = 1.0f / lsum;
        const int q = r + 8 * hi;
        abase[(size_t)ln * NTOK + q] = O0[r] * inv;
        if (ln < 8)
            abase[(size_t)(16 + ln) * NTOK + q] = O1[r] * inv;
    }
}

// ---------------------------------------------------------------------------
// Phase 3: out = wo[64x24] * attn + bo   (memory-trivial)
// ---------------------------------------------------------------------------
__global__ __launch_bounds__(256) void out_proj_kernel(
    const float* __restrict__ attn, const float* __restrict__ wo,
    const float* __restrict__ bo, float* __restrict__ out)
{
    __shared__ float sW[COUT * DH];
    __shared__ float sB[COUT];
    for (int t = threadIdx.x; t < COUT * DH; t += 256) sW[t] = wo[t];
    if (threadIdx.x < COUT) sB[threadIdx.x] = bo[threadIdx.x];
    __syncthreads();

    const int gid = blockIdx.x * 256 + threadIdx.x;
    const int b = gid / NTOK;
    const int i = gid - b * NTOK;

    float a[DH];
    const float* ap = attn + (size_t)b * DPAD * NTOK + i;
#pragma unroll
    for (int d = 0; d < DH; ++d) a[d] = ap[(size_t)d * NTOK];

    float* op = out + (size_t)b * COUT * NTOK + i;
    for (int c = 0; c < COUT; ++c) {
        float acc = sB[c];
        const float* w = sW + c * DH;
#pragma unroll
        for (int d = 0; d < DH; ++d) acc += w[d] * a[d];
        op[(size_t)c * NTOK] = acc;
    }
}

// ---------------------------------------------------------------------------
extern "C" void kernel_launch(void* const* d_in, const int* in_sizes, int n_in,
                              void* d_out, int out_size, void* d_ws, size_t ws_size,
                              hipStream_t stream) {
    const float* x  = (const float*)d_in[0];
    const float* wq = (const float*)d_in[1];
    const float* bq = (const float*)d_in[2];
    const float* wk = (const float*)d_in[3];
    const float* bk = (const float*)d_in[4];
    const float* wv = (const float*)d_in[5];
    const float* bv = (const float*)d_in[6];
    const float* wo = (const float*)d_in[7];
    const float* bo = (const float*)d_in[8];
    float* out = (float*)d_out;

    const size_t qkvElems = (size_t)BATCH * NTOK * DPAD;
    _Float16* Qh = (_Float16*)d_ws;
    _Float16* Kh = Qh + qkvElems;
    _Float16* Vh = Kh + qkvElems;
    float* attnb = (float*)(Vh + qkvElems);   // 64B-aligned (7,077,888B offset)

    qkv_proj_kernel<<<(BATCH * NTOK) / 256, 256, 0, stream>>>(
        x, wq, bq, wk, bk, wv, bv, Qh, Kh, Vh);
    flash_attn_kernel<<<(BATCH * (NTOK / 16)) / 8, 256, 0, stream>>>(
        Qh, Kh, Vh, attnb);
    out_proj_kernel<<<(BATCH * NTOK) / 256, 256, 0, stream>>>(
        attnb, wo, bo, out);
}
